// FocusScoreAttention_39822936768861
// MI455X (gfx1250) — matrix operand order repaired and checked by reading, not compiled, not run on hardware
//
#include <hip/hip_runtime.h>
#include <hip/hip_bf16.h>

// ---------------------------------------------------------------------------
// FocusScoreAttention on MI455X (gfx1250), wave32, bf16 WMMA w/ fp32 accum.
// B=8, S=2048, D=512. Focus bias f.f^T reduced to rank-1 closed form.
// A-tiles staged block-wide into LDS via TDM (tensor_load_to_lds); epilogue
// tiles DMA'd concurrently with the GEMM loop (in-order TENSORcnt).
// B-tiles streamed from global through a depth-4 rotating register buffer so
// ~6 b128 loads stay in flight per wave (hides L2 latency under the WMMAs).
// ---------------------------------------------------------------------------

#define BATCH 8
#define SEQ   2048
#define DM    512
#define SCALE_INV 0.04419417382415922f   // 1/sqrt(512)
#define LN_EPS 1e-5f

typedef __bf16 bf16_t;
typedef __attribute__((ext_vector_type(16))) __bf16 v16bf;
typedef __attribute__((ext_vector_type(8)))  __bf16 v8bf;
typedef __attribute__((ext_vector_type(8)))  float  v8f;

#if defined(__gfx1250__) && __has_builtin(__builtin_amdgcn_tensor_load_to_lds) \
    && __has_builtin(__builtin_amdgcn_s_wait_tensorcnt)
#define HAVE_TDM 1
#endif

#ifdef HAVE_TDM
typedef __attribute__((ext_vector_type(4))) unsigned int v4u;
typedef __attribute__((ext_vector_type(8))) int          v8i;
typedef __attribute__((ext_vector_type(4))) int          v4i;

__device__ __forceinline__ unsigned lds_addr_of(void* p) {
  return (unsigned)(unsigned long long)(__attribute__((address_space(3))) char*)(char*)p;
}

// 1-D contiguous TDM copy: nelem elements of (1<<dsz) bytes, global -> LDS.
// D# bitfields per CDNA5 ISA 8.3/8.4: group0 {count=1, lds_addr, global_addr,
// type=2}; group1 {data_size, tensor_dim0=nelem, tile_dim0=nelem, 1-D tile}.
__device__ __forceinline__ void tdm_load_1d(const void* gsrc, unsigned lds_addr,
                                            unsigned nelem, unsigned dsz) {
  unsigned long long ga = (unsigned long long)(size_t)gsrc;
  v4u g0;
  g0[0] = 1u;                                   // count=1 (valid), user mode
  g0[1] = lds_addr;                             // LDS byte address
  g0[2] = (unsigned)ga;                         // global_addr[31:0]
  g0[3] = (unsigned)((ga >> 32) & 0x01ffffffu)  // global_addr[56:32]
          | 0x80000000u;                        // type=2 ("image")
  v8i g1;
  g1[0] = (int)(dsz << 16);                     // wg_mask=0, data_size code
  g1[1] = (int)((nelem & 0xffffu) << 16);       // tensor_dim0[15:0]
  g1[2] = (int)((nelem >> 16) | (1u << 16));    // tensor_dim0[31:16], tdim1=1
  g1[3] = (int)((nelem & 0xffffu) << 16);       // tile_dim0 = nelem
  g1[4] = 0;                                    // tile_dim1=0 (1-D), tile_dim2=0
  g1[5] = (int)nelem;                           // tensor_dim0_stride[31:0]
  g1[6] = 0;
  g1[7] = 0;
  v4i z4 = {0, 0, 0, 0};
  v8i z8 = {0, 0, 0, 0, 0, 0, 0, 0};
  __builtin_amdgcn_tensor_load_to_lds(g0, g1, z4, z4, z8, 0);
}
#endif

// Stage a contiguous global tile into LDS. Under TDM only wave 0 issues the
// DMA (TDM ignores EXEC; one issue per wave). Fallback: cooperative copy.
__device__ __forceinline__ void stage_issue(void* lds_dst, const void* gsrc,
                                            unsigned nelem, unsigned dsz) {
#ifdef HAVE_TDM
  if ((threadIdx.x >> 5) == 0)
    tdm_load_1d(gsrc, lds_addr_of(lds_dst), nelem, dsz);
#else
  const uint4* s = (const uint4*)gsrc;
  uint4* d = (uint4*)lds_dst;
  unsigned n16 = (nelem << dsz) >> 4;
  for (unsigned i = threadIdx.x; i < n16; i += blockDim.x) d[i] = s[i];
#endif
}

// wait1: first of two in-order TDM ops complete; wait0: all done.
__device__ __forceinline__ void stage_wait1() {
#ifdef HAVE_TDM
  if ((threadIdx.x >> 5) == 0) __builtin_amdgcn_s_wait_tensorcnt(1);
#endif
  __syncthreads();
}
__device__ __forceinline__ void stage_wait0() {
#ifdef HAVE_TDM
  if ((threadIdx.x >> 5) == 0) __builtin_amdgcn_s_wait_tensorcnt(0);
#endif
  __syncthreads();
}

// D = A(16x32 bf16) * B(32x16 bf16) + C(16x16 f32)
__device__ __forceinline__ v8f wmma_bf16(v16bf a, v16bf b, v8f c) {
  return __builtin_amdgcn_wmma_f32_16x16x32_bf16(
      false, a, false, b, (short)0, c, false, false);
}

// A-matrix 16x32 bf16 (ISA 7.12.2): lanes 0-15 row M=lane hold K 0-7 / 16-23,
// lanes 16-31 hold K 8-15 / 24-31 => two contiguous 16B runs per lane.
// Works on LDS-resident tiles (addrspace inferred after inlining -> ds_load).
__device__ __forceinline__ v16bf load_a_bf(const bf16_t* base, int row0,
                                           int ld, int k0) {
  int lane = threadIdx.x & 31;
  int half = lane >> 4, r = lane & 15;
  const bf16_t* p = base + (size_t)(row0 + r) * ld + k0 + half * 8;
  v8bf lo = *(const v8bf*)p;
  v8bf hi = *(const v8bf*)(p + 16);
  v16bf out;
#pragma unroll
  for (int i = 0; i < 8; ++i) { out[i] = lo[i]; out[i + 8] = hi[i]; }
  return out;
}

// A-tile from fp32 source with inline f32->bf16 convert.
__device__ __forceinline__ v16bf load_a_f32(const float* base, int row0,
                                            int ld, int k0) {
  int lane = threadIdx.x & 31;
  int half = lane >> 4, r = lane & 15;
  const float* p = base + (size_t)(row0 + r) * ld + k0 + half * 8;
  v16bf out;
#pragma unroll
  for (int i = 0; i < 8; ++i) {
    out[i]     = (bf16_t)p[i];
    out[i + 8] = (bf16_t)p[16 + i];
  }
  return out;
}

// B-matrix 32x16 bf16: lanes 0-15 col N=lane hold K 0-7 / 16-23,
// lanes 16-31 hold K 8-15 / 24-31 => two contiguous 16B runs per lane.
// `base` is stored N-major with K contiguous (i.e. B^T row-major).
__device__ __forceinline__ v16bf load_b_bf(const bf16_t* base, int col0,
                                           int ld, int k0) {
  int lane = threadIdx.x & 31;
  int kg = lane >> 4, c = lane & 15;
  const bf16_t* p = base + (size_t)(col0 + c) * ld + k0 + kg * 8;
  v8bf lo = *(const v8bf*)p;
  v8bf hi = *(const v8bf*)(p + 16);
  v16bf out;
#pragma unroll
  for (int i = 0; i < 8; ++i) { out[i] = lo[i]; out[i + 8] = hi[i]; }
  return out;
}

// ---------------------------------------------------------------------------
// P1: focus scalars  a=||Wf||^2, b=Wf.bf, c=||bf||^2
// ---------------------------------------------------------------------------
__global__ __launch_bounds__(256) void k_prep_scalars(const float* Wf,
                                                      const float* bfv,
                                                      float* scal) {
  __shared__ float red[3][8];
  int t = threadIdx.x;
  float sa = 0.f, sb = 0.f, sc = 0.f;
  for (int i = t; i < DM; i += 256) {
    float w = Wf[i], b = bfv[i];
    sa += w * w; sb += w * b; sc += b * b;
  }
#pragma unroll
  for (int o = 16; o > 0; o >>= 1) {
    sa += __shfl_xor(sa, o, 32);
    sb += __shfl_xor(sb, o, 32);
    sc += __shfl_xor(sc, o, 32);
  }
  int wv = t >> 5, ln = t & 31;
  if (ln == 0) { red[0][wv] = sa; red[1][wv] = sb; red[2][wv] = sc; }
  __syncthreads();
  if (t == 0) {
    float A = 0.f, Bc = 0.f, C = 0.f;
    for (int i = 0; i < 8; ++i) { A += red[0][i]; Bc += red[1][i]; C += red[2][i]; }
    scal[0] = A; scal[1] = Bc; scal[2] = C;
  }
}

// ---------------------------------------------------------------------------
// P2: W [512x512] fp32 -> W^T bf16 (N-major, K contiguous) for B-matrix loads
// ---------------------------------------------------------------------------
__global__ __launch_bounds__(256) void k_transpose_w(const float* W, bf16_t* WT) {
  int idx = blockIdx.x * 256 + threadIdx.x;   // over 512*512
  int n = idx >> 9, k = idx & 511;
  WT[(size_t)n * DM + k] = (bf16_t)W[(size_t)k * DM + n];
}

// ---------------------------------------------------------------------------
// K1: fused q/k/v projection. 1024 blocks x 16 rows; 8 waves x 64 cols.
//     x tile (32KB fp32) TDM-staged; B tiles = 12-stream (4 n-tiles x 3
//     weights) through a depth-4 rotating buffer.
// ---------------------------------------------------------------------------
__global__ __launch_bounds__(256) void k_qkv(
    const float* x, const bf16_t* WqT, const bf16_t* WkT, const bf16_t* WvT,
    const float* bq, const float* bk, const float* bv,
    bf16_t* qb, bf16_t* kb, bf16_t* vT) {
  extern __shared__ char smem1[];
  float* xs = (float*)smem1;                       // 16*512 fp32 = 32KB
  int mbase = blockIdx.x * 16;
  stage_issue(xs, x + (size_t)mbase * DM, 16 * DM, 2);
  stage_wait0();

  int wv = threadIdx.x >> 5, lane = threadIdx.x & 31;
  int half = lane >> 4, nl = lane & 15;

  // tile index i in [0,12): weight wt = i%3 (q,k,v), n-tile nt = i/3
  auto bld = [&](int i, int kk) -> v16bf {
    const bf16_t* w = (i % 3 == 0) ? WqT : ((i % 3 == 1) ? WkT : WvT);
    return load_b_bf(w, wv * 64 + (i / 3) * 16, DM, kk);
  };

  v8f acc[12] = {};
  v16bf B[4];
#pragma unroll
  for (int i = 0; i < 4; ++i) B[i] = bld(i, 0);
  for (int k0 = 0; k0 < DM; k0 += 32) {
    v16bf a = load_a_f32(xs, 0, DM, k0);
#pragma unroll
    for (int i = 0; i < 12; ++i) {
      acc[i] = wmma_bf16(a, B[i & 3], acc[i]);
      int nn = i + 4;
      B[i & 3] = bld(nn % 12, k0 + 32 * (nn / 12));  // tail over-read: benign
    }
  }
#pragma unroll
  for (int nt = 0; nt < 4; ++nt) {
    int n = wv * 64 + nt * 16 + nl;
    float bqv = bq[n], bkv = bk[n], bvv = bv[n];
#pragma unroll
    for (int r = 0; r < 8; ++r) {
      int gr = mbase + r + 8 * half;          // global row in [0, B*S)
      size_t ro = (size_t)gr * DM + n;
      qb[ro] = (bf16_t)((acc[nt * 3 + 0][r] + bqv) * SCALE_INV);
      kb[ro] = (bf16_t)(acc[nt * 3 + 1][r] + bkv);
      int b = gr >> 11, s = gr & (SEQ - 1);
      vT[(size_t)b * DM * SEQ + (size_t)n * SEQ + s] =
          (bf16_t)(acc[nt * 3 + 2][r] + bvv);
    }
  }
}

// ---------------------------------------------------------------------------
// K2: scores = q.k^T (+ rank-1 focus bias) then row softmax.
//     q tile TDM-staged (wait tensorcnt<=1); fsk DMA overlapped with GEMM
//     (wait tensorcnt<=0 after). 16 query rows x full 2048 keys; scores tile
//     in 128KB LDS. 16 B-tiles per k-step via depth-4 rotation.
// ---------------------------------------------------------------------------
__global__ __launch_bounds__(256) void k_scores_softmax(
    const bf16_t* qb, const bf16_t* kb, const float* fs, const float* scal,
    float* attn_out, bf16_t* attn_bf) {
  extern __shared__ char smem2[];
  float*  sc  = (float*)smem2;              // 16*2048 fp32   (131072 B)
  float*  fsk = sc + 16 * SEQ;              // 2048 fp32      (8192 B)
  float*  fsq = fsk + SEQ;                  // 16 fp32
  bf16_t* qs  = (bf16_t*)(fsq + 16);        // 16*512 bf16    (16384 B)

  int b = blockIdx.x >> 7, mt = blockIdx.x & 127;
  int mbase = mt * 16;
  int t = threadIdx.x, wv = t >> 5, lane = t & 31;
  int half = lane >> 4, nl = lane & 15;

  const float*  fsb = fs + (size_t)b * SEQ;
  const bf16_t* qbb = qb + (size_t)b * SEQ * DM;
  const bf16_t* kbb = kb + (size_t)b * SEQ * DM;

  stage_issue(qs, qbb + (size_t)mbase * DM, 16 * DM, 1);  // A tile (first)
  stage_issue(fsk, fsb, SEQ, 2);                          // epilogue (second)
  if (t < 16) fsq[t] = fsb[mbase + t];
  stage_wait1();                     // q tile ready (in-order completion)

  float fa = scal[0], fb = scal[1], fc = scal[2];
  int ncol0 = wv * 256;
  v8f acc[16] = {};
  v16bf B[4];
#pragma unroll
  for (int i = 0; i < 4; ++i) B[i] = load_b_bf(kbb, ncol0 + i * 16, DM, 0);
  for (int k0 = 0; k0 < DM; k0 += 32) {
    v16bf a = load_a_bf(qs, 0, DM, k0);
#pragma unroll
    for (int i = 0; i < 16; ++i) {
      acc[i] = wmma_bf16(a, B[i & 3], acc[i]);
      int nn = i + 4;
      B[i & 3] = load_b_bf(kbb, ncol0 + (nn & 15) * 16, DM,
                           k0 + 32 * (nn >> 4));          // tail over-read ok
    }
  }
  stage_wait0();                     // fsk DMA overlapped with GEMM, now done

#pragma unroll
  for (int nt = 0; nt < 16; ++nt) {
    int n = ncol0 + nt * 16 + nl;
    float fkn = fsk[n];
#pragma unroll
    for (int r = 0; r < 8; ++r) {
      int m = r + 8 * half;
      float fqm = fsq[m];
      sc[m * SEQ + n] = acc[nt][r] + fa * fqm * fkn + fb * (fqm + fkn) + fc;
    }
  }
  __syncthreads();

  // softmax: each wave owns 2 rows; wave32 shuffle reductions
#pragma unroll
  for (int rr = 0; rr < 2; ++rr) {
    int m = wv * 2 + rr;
    float* row = sc + m * SEQ;
    float mx = -3.4e38f;
    for (int i = lane; i < SEQ; i += 32) mx = fmaxf(mx, row[i]);
#pragma unroll
    for (int o = 16; o > 0; o >>= 1) mx = fmaxf(mx, __shfl_xor(mx, o, 32));
    float sum = 0.f;
    for (int i = lane; i < SEQ; i += 32) {
      float e = __expf(row[i] - mx);
      row[i] = e; sum += e;
    }
#pragma unroll
    for (int o = 16; o > 0; o >>= 1) sum += __shfl_xor(sum, o, 32);
    float inv = 1.0f / sum;
    size_t orow = ((size_t)b * SEQ + mbase + m) * SEQ;
    float*  ao = attn_out + orow;
    bf16_t* ab = attn_bf + orow;
    for (int i = lane; i < SEQ; i += 32) {
      float v = row[i] * inv;
      ao[i] = v;                 // fp32 attn output (returned tensor)
      ab[i] = (bf16_t)v;         // bf16 staging for attn @ v
    }
  }
}

// ---------------------------------------------------------------------------
// K3: attended = attn @ v  (M=S, N=D, K=S per batch).
//     64KB attn A-tile TDM-staged once per block; 4 B-tiles rotated with
//     prefetch distance 4 (pipeline crosses the k-step boundary).
// ---------------------------------------------------------------------------
__global__ __launch_bounds__(256) void k_attnv(const bf16_t* attn_bf,
                                               const bf16_t* vT,
                                               bf16_t* att_out) {
  extern __shared__ char smem3[];
  bf16_t* as = (bf16_t*)smem3;              // 16*2048 bf16 = 64KB
  int b = blockIdx.x >> 7, mt = blockIdx.x & 127;
  int mbase = mt * 16;
  const bf16_t* ab = attn_bf + (size_t)b * SEQ * SEQ;
  const bf16_t* vb = vT + (size_t)b * DM * SEQ;

  stage_issue(as, ab + (size_t)mbase * SEQ, 16 * SEQ, 1);
  stage_wait0();

  int wv = threadIdx.x >> 5, lane = threadIdx.x & 31;
  int half = lane >> 4, nl = lane & 15;
  int c0 = wv * 64;
  v8f acc[4] = {};
  v16bf B[4];
#pragma unroll
  for (int i = 0; i < 4; ++i) B[i] = load_b_bf(vb, c0 + i * 16, SEQ, 0);
  for (int k0 = 0; k0 < SEQ; k0 += 32) {
    v16bf a = load_a_bf(as, 0, SEQ, k0);
#pragma unroll
    for (int i = 0; i < 4; ++i) {
      acc[i] = wmma_bf16(a, B[i], acc[i]);
      B[i] = load_b_bf(vb, c0 + i * 16, SEQ, k0 + 32);    // tail over-read ok
    }
  }
#pragma unroll
  for (int nt = 0; nt < 4; ++nt) {
    int n = c0 + nt * 16 + nl;
#pragma unroll
    for (int r = 0; r < 8; ++r) {
      int gr = b * SEQ + mbase + r + 8 * half;
      att_out[(size_t)gr * DM + n] = (bf16_t)acc[nt][r];
    }
  }
}

// ---------------------------------------------------------------------------
// K4: out = LayerNorm(x + attended @ Wo + bo) * gamma + beta  (fused).
//     attended A-tile staged for the GEMM; x residual tile DMA'd
//     concurrently with the GEMM loop. Rotated B-tile prefetch.
// ---------------------------------------------------------------------------
__global__ __launch_bounds__(256) void k_out_ln(
    const bf16_t* att, const bf16_t* WoT, const float* bo, const float* x,
    const float* gamma, const float* beta, float* out) {
  extern __shared__ char smem4[];
  float*  hbuf = (float*)smem4;             // 16*512 fp32 = 32KB
  float*  xs   = hbuf + 16 * DM;            // 16*512 fp32 = 32KB
  bf16_t* atts = (bf16_t*)(xs + 16 * DM);   // 16*512 bf16 = 16KB

  int mbase = blockIdx.x * 16;
  stage_issue(atts, att + (size_t)mbase * DM, 16 * DM, 1);  // A tile (first)
  stage_issue(xs, x + (size_t)mbase * DM, 16 * DM, 2);      // residual (second)
  stage_wait1();

  int t = threadIdx.x, wv = t >> 5, lane = t & 31;
  int half = lane >> 4, nl = lane & 15;
  int c0 = wv * 64;
  v8f acc[4] = {};
  v16bf B[4];
#pragma unroll
  for (int i = 0; i < 4; ++i) B[i] = load_b_bf(WoT, c0 + i * 16, DM, 0);
  for (int k0 = 0; k0 < DM; k0 += 32) {
    v16bf a = load_a_bf(atts, 0, DM, k0);
#pragma unroll
    for (int i = 0; i < 4; ++i) {
      acc[i] = wmma_bf16(a, B[i], acc[i]);
      B[i] = load_b_bf(WoT, c0 + i * 16, DM, k0 + 32);     // tail over-read ok
    }
  }
  stage_wait0();                 // x residual DMA overlapped with GEMM

#pragma unroll
  for (int nt = 0; nt < 4; ++nt) {
    int n = c0 + nt * 16 + nl;
    float bon = bo[n];
#pragma unroll
    for (int r = 0; r < 8; ++r) {
      int m = r + 8 * half;
      hbuf[m * DM + n] = acc[nt][r] + bon + xs[m * DM + n];
    }
  }
  __syncthreads();
#pragma unroll
  for (int rr = 0; rr < 2; ++rr) {
    int m = wv * 2 + rr;
    float* row = hbuf + m * DM;
    float s = 0.f;
    for (int i = lane; i < DM; i += 32) s += row[i];
#pragma unroll
    for (int o = 16; o > 0; o >>= 1) s += __shfl_xor(s, o, 32);
    float mu = s * (1.0f / DM);
    float v = 0.f;
    for (int i = lane; i < DM; i += 32) { float d = row[i] - mu; v += d * d; }
#pragma unroll
    for (int o = 16; o > 0; o >>= 1) v += __shfl_xor(v, o, 32);
    float rs = rsqrtf(v * (1.0f / DM) + LN_EPS);
    float* op = out + (size_t)(mbase + m) * DM;
    for (int i = lane; i < DM; i += 32)
      op[i] = (row[i] - mu) * rs * gamma[i] + beta[i];
  }
}

// ---------------------------------------------------------------------------
extern "C" void kernel_launch(void* const* d_in, const int* in_sizes, int n_in,
                              void* d_out, int out_size, void* d_ws,
                              size_t ws_size, hipStream_t stream) {
  const float* x     = (const float*)d_in[0];
  const float* fs    = (const float*)d_in[1];
  const float* Wq    = (const float*)d_in[2];
  const float* bq    = (const float*)d_in[3];
  const float* Wk    = (const float*)d_in[4];
  const float* bk    = (const float*)d_in[5];
  const float* Wv    = (const float*)d_in[6];
  const float* bv    = (const float*)d_in[7];
  const float* Wf    = (const float*)d_in[8];
  const float* bfv   = (const float*)d_in[9];
  const float* Wo    = (const float*)d_in[10];
  const float* bo    = (const float*)d_in[11];
  const float* gamma = (const float*)d_in[12];
  const float* beta  = (const float*)d_in[13];

  char* ws = (char*)d_ws;
  float*  scal = (float*)ws;                       // 3 floats
  bf16_t* WqT  = (bf16_t*)(ws + 1024);             // 4 x 512KB transposed W
  bf16_t* WkT  = WqT + (size_t)DM * DM;
  bf16_t* WvT  = WkT + (size_t)DM * DM;
  bf16_t* WoT  = WvT + (size_t)DM * DM;
  bf16_t* qb    = (bf16_t*)(ws + ((size_t)4 << 20));        // 16 MB
  bf16_t* kb    = qb  + (size_t)BATCH * SEQ * DM;           // 16 MB
  bf16_t* vT    = kb  + (size_t)BATCH * SEQ * DM;           // 16 MB
  bf16_t* att   = vT  + (size_t)BATCH * SEQ * DM;           // 16 MB
  bf16_t* attnb = att + (size_t)BATCH * SEQ * DM;           // 64 MB

  float* out      = (float*)d_out;                          // [B,S,D]
  float* attn_out = out + (size_t)BATCH * SEQ * DM;         // [B,S,S]

  k_prep_scalars<<<1, 256, 0, stream>>>(Wf, bfv, scal);
  k_transpose_w<<<1024, 256, 0, stream>>>(Wq, WqT);
  k_transpose_w<<<1024, 256, 0, stream>>>(Wk, WkT);
  k_transpose_w<<<1024, 256, 0, stream>>>(Wv, WvT);
  k_transpose_w<<<1024, 256, 0, stream>>>(Wo, WoT);

  size_t sm1 = (size_t)16 * DM * 4;                          // 32 KB
  k_qkv<<<BATCH * SEQ / 16, 256, sm1, stream>>>(x, WqT, WkT, WvT, bq, bk, bv,
                                                qb, kb, vT);

  size_t sm2 = (size_t)(16 * SEQ + SEQ + 16) * 4 + 16 * DM * 2;  // ~152 KB
  k_scores_softmax<<<BATCH * (SEQ / 16), 256, sm2, stream>>>(
      qb, kb, fs, scal, attn_out, attnb);

  size_t sm3 = (size_t)16 * SEQ * 2;                         // 64 KB
  k_attnv<<<BATCH * (SEQ / 16), 256, sm3, stream>>>(attnb, vT, att);

  size_t sm4 = (size_t)16 * DM * 4 * 2 + 16 * DM * 2;        // 80 KB
  k_out_ln<<<BATCH * SEQ / 16, 256, sm4, stream>>>(att, WoT, bo, x, gamma,
                                                   beta, out);
}
